// Attention_56513179681272
// MI455X (gfx1250) — matrix operand order repaired and checked by reading, not compiled, not run on hardware
//
#include <hip/hip_runtime.h>
#include <hip/hip_bf16.h>
#include <math.h>

// Problem constants (from reference): B=32, T=8192, F=256, H=128
#define B_  32
#define T_  8192
#define F_  256
#define H_  128
#define R_  (B_ * T_)            // 262144 flattened rows (b,t)
#define ROWS_PER_BLOCK 64        // 4 waves x 16 rows
#define NWAVES 4
#define KC_ (F_ / 32)            // 8 K-chunks of 32 (bf16 WMMA K)
#define NC_ (H_ / 16)            // 8 N-tiles of 16

typedef __attribute__((ext_vector_type(16))) __bf16        v16bf;
typedef __attribute__((ext_vector_type(8)))  float         v8f;
typedef __attribute__((ext_vector_type(4)))  unsigned int  v4u;
typedef __attribute__((ext_vector_type(8)))  int           v8i;
typedef __attribute__((ext_vector_type(4)))  int           v4i;

__device__ inline unsigned pack_bf16(float lo, float hi) {
  union { __bf16 b[2]; unsigned u; } cv;
  cv.b[0] = (__bf16)lo;
  cv.b[1] = (__bf16)hi;
  return cv.u;
}

// ---------------------------------------------------------------------------
// Kernel 1: bias[b,h] = dec_output[b,:] . W2[:,h]   (32x128, trivial)
// ---------------------------------------------------------------------------
__global__ void k_bias(const float* __restrict__ dec,
                       const float* __restrict__ W2,
                       float* __restrict__ bias) {
  const int b = blockIdx.x;      // 32 blocks
  const int h = threadIdx.x;     // 128 threads
  float s = 0.f;
  #pragma unroll 4
  for (int f = 0; f < F_; ++f)
    s += dec[b * F_ + f] * W2[f * H_ + h];
  bias[b * H_ + h] = s;
}

// ---------------------------------------------------------------------------
// Kernel 2: pre-swizzle W1 (fp32 [F][H]) into CDNA5 bf16 B-fragment layout.
// Fragment (kc,nc): 32x16 bf16 tile of W1. B layout used here (16-bit, 32x16, contiguous k order):
//   lane<16 : N=lane,    K = i        (i = packed element index 0..15)
//   lane>=16: N=lane-16, K = 16 + i
// Stored as [kc][nc][lane][8] u32 so each consumer lane reads 32 contiguous B.
// ---------------------------------------------------------------------------
__global__ void k_w1frag(const float* __restrict__ W1,
                         unsigned* __restrict__ frag) {
  const int t    = blockIdx.x * blockDim.x + threadIdx.x;  // 0..2047
  const int lane = t & 31;
  const int nc   = (t >> 5) & 7;
  const int kc   = t >> 8;
  const int h    = nc * 16 + (lane & 15);
  const int kb   = (lane < 16) ? 0 : 16;
  #pragma unroll
  for (int j = 0; j < 8; ++j) {
    const int f0 = kc * 32 + kb + 2 * j;
    frag[t * 8 + j] = pack_bf16(W1[f0 * H_ + h], W1[(f0 + 1) * H_ + h]);
  }
}

// ---------------------------------------------------------------------------
// Main kernel: per block of 64 rows:
//   TDM: global fp32 encoder tile (64x256 = 64KB) -> LDS
//   per wave: 16 rows -> 8(K) x 8(N) v_wmma_f32_16x16x32_bf16 -> tanh -> .V
// ---------------------------------------------------------------------------
__global__ __launch_bounds__(NWAVES * 32) void attn_main(
    const float*    __restrict__ enc,
    const unsigned* __restrict__ w1frag,
    const float*    __restrict__ bias,
    const float*    __restrict__ V,
    float*          __restrict__ out) {
  __shared__ float sA[ROWS_PER_BLOCK * F_];   // 64 KB fp32 encoder tile

  const int lane = threadIdx.x & 31;
  const int wv   = threadIdx.x >> 5;
  const long long rowStart = (long long)blockIdx.x * ROWS_PER_BLOCK;
  const int b = (int)(rowStart / T_);         // 64 | T, so b uniform per block

  // ---- Stage encoder tile into LDS with the Tensor Data Mover (wave 0) ----
#if __has_builtin(__builtin_amdgcn_tensor_load_to_lds)
  if (threadIdx.x < 32) {
    const unsigned long long ga =
        (unsigned long long)(const void*)(enc + rowStart * F_);
    const unsigned ldsa = (unsigned)(unsigned long long)(void*)&sA[0];
    v4u g0;
    g0[0] = 1u;                                   // count=1 valid descriptor
    g0[1] = ldsa;                                 // lds_addr
    g0[2] = (unsigned)(ga & 0xffffffffu);         // global_addr[31:0]
    g0[3] = (unsigned)(ga >> 32) | (2u << 30);    // global_addr[56:32]|type=2
    v8i g1;
    g1[0] = (int)(2u << 16);                      // data_size=2 (4 bytes)
    g1[1] = (int)((unsigned)(F_ & 0xffff) << 16); // tensor_dim0[15:0]=256
    g1[2] = (int)((F_ >> 16) | ((R_ & 0xffff) << 16));   // dim0 hi | dim1 lo
    g1[3] = (int)((R_ >> 16) | ((unsigned)F_ << 16));    // dim1 hi | tile_dim0
    g1[4] = ROWS_PER_BLOCK;                       // tile_dim1=64, tile_dim2=0
    g1[5] = F_;                                   // tensor_dim0_stride=256
    g1[6] = 0;
    g1[7] = 0;
    v4i z = (v4i)0;
#if defined(__clang_major__) && (__clang_major__ >= 23)
    __builtin_amdgcn_tensor_load_to_lds(g0, g1, z, z, (v8i)0, 0);
#else
    __builtin_amdgcn_tensor_load_to_lds(g0, g1, z, z, 0);
#endif
    __builtin_amdgcn_s_wait_tensorcnt(0);
  }
#else
  // Fallback: cooperative coalesced load
  for (int i = threadIdx.x; i < ROWS_PER_BLOCK * F_; i += NWAVES * 32)
    sA[i] = enc[rowStart * F_ + i];
#endif
  __syncthreads();

  // ---- WMMA: D[16x16] tiles, A = 16x32 bf16 per K-chunk ----
  // A layout (16-bit 16x32): lane<16: M=lane, K = i (i = element 0..15);
  // lane>=16: M=lane-16, K = 16 + i (matches B fragment order).
  const int m  = lane & 15;
  const int kb = (lane < 16) ? 0 : 16;
  const float* aRow = &sA[(wv * 16 + m) * F_];

  v8f c[NC_];
  #pragma unroll
  for (int nc = 0; nc < NC_; ++nc) {
    #pragma unroll
    for (int r = 0; r < 8; ++r) c[nc][r] = 0.f;
  }

  #pragma unroll
  for (int kc = 0; kc < KC_; ++kc) {
    v16bf a;
    #pragma unroll
    for (int j = 0; j < 8; ++j) {
      const int K0 = 2 * j + kb;
      a[2 * j]     = (__bf16)aRow[kc * 32 + K0];
      a[2 * j + 1] = (__bf16)aRow[kc * 32 + K0 + 1];
    }
    const v16bf* bf = (const v16bf*)w1frag + (size_t)kc * NC_ * 32;
    #pragma unroll
    for (int nc = 0; nc < NC_; ++nc) {
      v16bf bb = bf[nc * 32 + lane];   // 32B/lane contiguous, L2-resident
      c[nc] = __builtin_amdgcn_wmma_f32_16x16x32_bf16(
          false, a, false, bb, (short)0, c[nc], false, false);
    }
  }

  // ---- Epilogue: score[m] = sum_h V[h] * tanh(c[m][h] + bias[b][h]) ----
  // C layout: VGPR r -> M = r + (lane<16 ? 0 : 8); N = lane&15.
  float acc[8];
  #pragma unroll
  for (int r = 0; r < 8; ++r) acc[r] = 0.f;
  #pragma unroll
  for (int nc = 0; nc < NC_; ++nc) {
    const int h  = nc * 16 + (lane & 15);
    const float bh = bias[b * H_ + h];
    const float vh = V[h];
    #pragma unroll
    for (int r = 0; r < 8; ++r)
      acc[r] += vh * tanhf(c[nc][r] + bh);
  }
  // Reduce over the 16 lanes of each half-wave (h-partials)
  #pragma unroll
  for (int off = 1; off < 16; off <<= 1) {
    #pragma unroll
    for (int r = 0; r < 8; ++r)
      acc[r] += __shfl_xor(acc[r], off, 32);
  }
  if ((lane & 15) == 0) {
    const int mbase = (lane < 16) ? 0 : 8;
    const long long rw = rowStart + wv * 16 + mbase;
    #pragma unroll
    for (int r = 0; r < 8; ++r) out[rw + r] = acc[r];
  }
}

// ---------------------------------------------------------------------------
extern "C" void kernel_launch(void* const* d_in, const int* in_sizes, int n_in,
                              void* d_out, int out_size, void* d_ws, size_t ws_size,
                              hipStream_t stream) {
  const float* enc = (const float*)d_in[0];  // [B,T,F]
  const float* dec = (const float*)d_in[1];  // [B,F]
  const float* W1  = (const float*)d_in[2];  // [F,H]
  const float* W2  = (const float*)d_in[3];  // [F,H]
  const float* V   = (const float*)d_in[4];  // [H,1]
  float* out = (float*)d_out;                // [B,T]

  float*    bias   = (float*)d_ws;                                   // 16 KB
  unsigned* w1frag = (unsigned*)((char*)d_ws + B_ * H_ * sizeof(float)); // 64 KB

  k_bias<<<B_, H_, 0, stream>>>(dec, W2, bias);
  k_w1frag<<<(KC_ * NC_ * 32) / 128, 128, 0, stream>>>(W1, w1frag);
  attn_main<<<R_ / ROWS_PER_BLOCK, NWAVES * 32, 0, stream>>>(
      enc, w1frag, bias, V, out);
}
